// VRWKV_SpatialMix_86912958202223
// MI455X (gfx1250) — matrix operand order repaired and checked by reading, not compiled, not run on hardware
//
#include <hip/hip_runtime.h>

// ---------------------------------------------------------------------------
// VRWKV SpatialMix for MI455X (gfx1250, wave32, WMMA + async LDS staging)
// B=16, H=W=32, T=1024, C=768, Hd=64, V=3  (hardcoded to the reference shapes)
//
// GEMMs: bf16 WMMA (v_wmma_f32_16x16x32_bf16). Block = 8 waves = 256x64 tile
// (32x64 per wave). B-tile staged in LDS via GLOBAL_LOAD_ASYNC_TO_LDS_B128
// (ASYNCcnt), double buffered; A fragments software-pipelined in registers.
// ---------------------------------------------------------------------------

typedef __attribute__((ext_vector_type(16))) __bf16 v16bf;
typedef __attribute__((ext_vector_type(8)))  float  v8f;
typedef int v4i __attribute__((__vector_size__(16)));

union ABfrag { uint4 u[2]; v16bf v; };

static constexpr int Bb = 16;
static constexpr int Hh_ = 32;
static constexpr int Ww_ = 32;
static constexpr int Tt = Hh_ * Ww_;          // 1024
static constexpr int Cc = 768;
static constexpr int Mtok = Bb * Tt;          // 16384

#if defined(__gfx1250__) && \
    __has_builtin(__builtin_amdgcn_global_load_async_to_lds_b128) && \
    __has_builtin(__builtin_amdgcn_s_wait_asynccnt)
#define USE_ASYNC_LDS 1
#else
#define USE_ASYNC_LDS 0
#endif

// ---- fp32 <-> bf16 helpers (round-to-nearest-even), stored as ushort ----
__device__ __forceinline__ unsigned short f2bf(float f) {
  unsigned int u = __float_as_uint(f);
  u += 0x7FFFu + ((u >> 16) & 1u);
  return (unsigned short)(u >> 16);
}
__device__ __forceinline__ float bf2f(unsigned short s) {
  return __uint_as_float(((unsigned int)s) << 16);
}

// ---------------------------------------------------------------------------
// fp32 -> bf16 convert (one element per thread)
// ---------------------------------------------------------------------------
__global__ __launch_bounds__(256) void cvt_f32_bf16_kernel(
    const float* __restrict__ in, unsigned short* __restrict__ out, int n) {
  int i = blockIdx.x * 256 + threadIdx.x;
  if (i < n) out[i] = f2bf(in[i]);
}

// concat three 64-float bias vectors -> 192
__global__ void concat_bias_kernel(const float* a, const float* b,
                                   const float* c, float* o) {
  int i = threadIdx.x;
  if (i < 64) { o[i] = a[i]; o[64 + i] = b[i]; o[128 + i] = c[i]; }
}

// ---------------------------------------------------------------------------
// Tiled bf16 WMMA GEMM:  Out[M,N] = A[M,K] * W[N,K]^T   (+ epilogue)
//   EPI 0: fp32 store | EPI 1: +bias, ReLU, fp32 | EPI 2: sigmoid, bf16
//
// Block (256 thr / 8 waves) computes 256x64: wave w owns rows m0..m0+31
// (two 16-row A fragments), all waves share columns n0..n0+63.
// Per K-step of 32:
//   - 64x32 bf16 B tile staged in LDS (async double buffer), 40-element
//     padded row stride -> ds_load_b128 bank starts {0,20,40,...}: conflict
//     free across a half-wave.
//   - all 4 B fragments pulled to registers first (one dscnt wait), then
//     8 independent WMMAs issued back-to-back.
//   - next K-step's A fragments prefetched from global before the WMMA block.
// Fragment lane layouts per CDNA5 ISA 7.12.2 (16-bit A 16x32 / B 32x16).
// ---------------------------------------------------------------------------
template <int EPI>
__global__ __launch_bounds__(256) void gemm_bf16_wmma_kernel(
    const unsigned short* __restrict__ A, const unsigned short* __restrict__ W,
    const float* __restrict__ bias, void* __restrict__ Out,
    int M, int N, int K) {
  constexpr int BKP = 40;                       // padded row stride (elems)
  __shared__ unsigned short Bt[2][64 * BKP];    // 10.2 KB

  const int tid  = threadIdx.x;
  const int lane = tid & 31;
  const int wid  = tid >> 5;
  const int nblk = N >> 6;
  const int bm   = blockIdx.x / nblk;
  const int bn   = blockIdx.x % nblk;
  const int m0   = bm * 256 + wid * 32;
  const int n0   = bn * 64;

  const int half = lane >> 4;                   // 0: lanes 0-15, 1: 16-31
  const int l16  = lane & 15;

  // cooperative B-tile fill: thread -> (row frow, 8-elem chunk fchk)
  const int frow = tid >> 2;
  const int fchk = (tid & 3) * 8;
  const unsigned short* wsrc = W + (size_t)(n0 + frow) * K + fchk;
  unsigned short* ldst[2] = { &Bt[0][frow * BKP + fchk],
                              &Bt[1][frow * BKP + fchk] };

  const unsigned short* arow0 = A + (size_t)(m0 + l16) * K + (half ? 8 : 0);
  const unsigned short* arow1 = arow0 + (size_t)16 * K;

  auto fill = [&](int bufi, int kk) {
#if USE_ASYNC_LDS
    __builtin_amdgcn_global_load_async_to_lds_b128(
        (__attribute__((address_space(1))) v4i*)(uintptr_t)(wsrc + kk),
        (__attribute__((address_space(3))) v4i*)(unsigned int)(uintptr_t)ldst[bufi],
        0, 0);
#else
    *(uint4*)ldst[bufi] = *(const uint4*)(wsrc + kk);
#endif
  };
  auto fill_wait = [&]() {
#if USE_ASYNC_LDS
    __builtin_amdgcn_s_wait_asynccnt(0);
#endif
    __syncthreads();
  };

  const int NS = K >> 5;                        // 24 K-steps
  v8f acc[2][4] = {};

  fill(0, 0);
  fill_wait();

  ABfrag a_cur[2], a_nxt[2];
  a_cur[0].u[0] = *(const uint4*)(arow0);
  a_cur[0].u[1] = *(const uint4*)(arow0 + 16);
  a_cur[1].u[0] = *(const uint4*)(arow1);
  a_cur[1].u[1] = *(const uint4*)(arow1 + 16);

#pragma unroll 2
  for (int s = 0; s < NS; ++s) {
    const int kk = s << 5;
    if (s + 1 < NS) {
      fill((s + 1) & 1, kk + 32);               // overlaps with WMMAs below
      a_nxt[0].u[0] = *(const uint4*)(arow0 + kk + 32);
      a_nxt[0].u[1] = *(const uint4*)(arow0 + kk + 48);
      a_nxt[1].u[0] = *(const uint4*)(arow1 + kk + 32);
      a_nxt[1].u[1] = *(const uint4*)(arow1 + kk + 48);
    }

    const unsigned short* bbase = Bt[s & 1];
    ABfrag b[4];
#pragma unroll
    for (int j = 0; j < 4; ++j) {
      const unsigned short* bl = bbase + (j * 16 + l16) * BKP + half * 8;
      b[j].u[0] = *(const uint4*)(bl);
      b[j].u[1] = *(const uint4*)(bl + 16);
    }
#pragma unroll
    for (int g = 0; g < 2; ++g)
#pragma unroll
      for (int j = 0; j < 4; ++j)
        acc[g][j] = __builtin_amdgcn_wmma_f32_16x16x32_bf16(
            false, a_cur[g].v, false, b[j].v, (short)0, acc[g][j], false, false);

    if (s + 1 < NS) {
      fill_wait();                              // fill(s+1) done + readers sync
      a_cur[0] = a_nxt[0];
      a_cur[1] = a_nxt[1];
    }
  }

  // C/D layout: VGPR i -> row m0 + g*16 + i + half*8 ; col n0 + j*16 + l16
#pragma unroll
  for (int g = 0; g < 2; ++g) {
#pragma unroll
    for (int j = 0; j < 4; ++j) {
      const int col = n0 + j * 16 + l16;
#pragma unroll
      for (int i = 0; i < 8; ++i) {
        const int row = m0 + g * 16 + i + half * 8;
        float v = acc[g][j][i];
        size_t idx = (size_t)row * N + col;
        if (EPI == 1) {
          ((float*)Out)[idx] = fmaxf(v + bias[col], 0.0f);
        } else if (EPI == 2) {
          ((unsigned short*)Out)[idx] = f2bf(1.0f / (1.0f + __expf(-v)));
        } else {
          ((float*)Out)[idx] = v;
        }
      }
    }
  }
}

// ---------------------------------------------------------------------------
// sema softmax weights: per (token, sema): logits[v] = b2[v] + h(64) . w2[v]
// ---------------------------------------------------------------------------
__global__ __launch_bounds__(256) void sema_softmax_kernel(
    const float* __restrict__ Hbuf,
    const float* w2k, const float* b2k, const float* w2v, const float* b2v,
    const float* w2r, const float* b2r, float* __restrict__ wgt) {
  int idx = blockIdx.x * 256 + threadIdx.x;       // token*3 + s
  if (idx >= Mtok * 3) return;
  int s = idx % 3, tok = idx / 3;
  const float* hrow = Hbuf + (size_t)tok * 192 + s * 64;
  const float* w2 = (s == 0) ? w2k : (s == 1) ? w2v : w2r;
  const float* b2 = (s == 0) ? b2k : (s == 1) ? b2v : b2r;
  float lg[3];
#pragma unroll
  for (int vv = 0; vv < 3; ++vv) {
    float acc = b2[vv];
    for (int hh = 0; hh < 64; ++hh) acc += hrow[hh] * w2[vv * 64 + hh];
    lg[vv] = acc;
  }
  float m = fmaxf(lg[0], fmaxf(lg[1], lg[2]));
  float e0 = __expf(lg[0] - m), e1 = __expf(lg[1] - m), e2 = __expf(lg[2] - m);
  float inv = 1.0f / (e0 + e1 + e2);
  float* o = wgt + (size_t)tok * 9 + s * 3;
  o[0] = e0 * inv; o[1] = e1 * inv; o[2] = e2 * inv;
}

// ---------------------------------------------------------------------------
// shift-mix: xx_s = sum_r wgt[s,r] * q_shift(x, r); x*_b = bf16(x*mix + xx*(1-mix))
// ---------------------------------------------------------------------------
__global__ __launch_bounds__(256) void shiftmix_kernel(
    const float* __restrict__ x, const float* __restrict__ wgt,
    const float* __restrict__ mk, const float* __restrict__ mv,
    const float* __restrict__ mr,
    unsigned short* __restrict__ xk, unsigned short* __restrict__ xv,
    unsigned short* __restrict__ xr) {
  size_t e = (size_t)blockIdx.x * 256 + threadIdx.x;
  if (e >= (size_t)Mtok * Cc) return;
  int c = (int)(e % Cc);
  size_t bt = e / Cc;
  int t = (int)(bt % Tt);
  int b = (int)(bt / Tt);
  int h0 = t >> 5, w0 = t & 31;
  int grp = c / 192;
  const float* xbase = x + (size_t)b * Tt * Cc;

  float vals[3];
#pragma unroll
  for (int r = 1; r <= 3; ++r) {
    int hs = h0, ws = w0;
    bool ok;
    if (grp == 0)      { ws = w0 - r; ok = ws >= 0;  }
    else if (grp == 1) { ws = w0 + r; ok = ws < Ww_; }
    else if (grp == 2) { hs = h0 - r; ok = hs >= 0;  }
    else               { hs = h0 + r; ok = hs < Hh_; }
    vals[r - 1] = ok ? xbase[(size_t)(hs * Ww_ + ws) * Cc + c] : 0.0f;
  }
  const float* wg = wgt + bt * 9;
  float xc  = xbase[(size_t)t * Cc + c];
  float xxk = wg[0] * vals[0] + wg[1] * vals[1] + wg[2] * vals[2];
  float xxv = wg[3] * vals[0] + wg[4] * vals[1] + wg[5] * vals[2];
  float xxr = wg[6] * vals[0] + wg[7] * vals[1] + wg[8] * vals[2];
  float a;
  a = mk[c]; xk[e] = f2bf(xc * a + xxk * (1.0f - a));
  a = mv[c]; xv[e] = f2bf(xc * a + xxv * (1.0f - a));
  a = mr[c]; xr[e] = f2bf(xc * a + xxr * (1.0f - a));
}

// ---------------------------------------------------------------------------
// WKV scan (stable max trick). One thread per (b, c); coalesced over c.
// 1-step load-ahead so the next (k,v) global loads overlap the exp chain.
// ---------------------------------------------------------------------------
__global__ __launch_bounds__(256) void wkv_kernel(
    const float* __restrict__ k, const float* __restrict__ v,
    const float* __restrict__ decay, const float* __restrict__ first,
    float* __restrict__ y) {
  int idx = blockIdx.x * 256 + threadIdx.x;       // b*C + c
  if (idx >= Bb * Cc) return;
  int b = idx / Cc, c = idx % Cc;
  const float w = decay[c] / (float)Tt;
  const float u = first[c] / (float)Tt;
  const float* kp = k + (size_t)b * Tt * Cc + c;
  const float* vp = v + (size_t)b * Tt * Cc + c;
  float* yp = y + (size_t)b * Tt * Cc + c;
  float p = 0.0f, q = 0.0f, o = -1e38f;
  float kt = kp[0], vt = vp[0];
  for (int t = 0; t < Tt; ++t) {
    float ktn = 0.0f, vtn = 0.0f;
    if (t + 1 < Tt) {
      ktn = kp[(size_t)(t + 1) * Cc];
      vtn = vp[(size_t)(t + 1) * Cc];
    }
    float no = fmaxf(o, u + kt);
    float Aa = __expf(o - no);
    float Bc = __expf(u + kt - no);
    yp[(size_t)t * Cc] = (Aa * p + Bc * vt) / (Aa * q + Bc);
    float no2 = fmaxf(w + o, kt);
    float A2 = __expf(w + o - no2);
    float B2 = __expf(kt - no2);
    p = A2 * p + B2 * vt;
    q = A2 * q + B2;
    o = no2;
    kt = ktn; vt = vtn;
  }
}

// ---------------------------------------------------------------------------
// LayerNorm over C + sigmoid gate (gate precomputed bf16) -> bf16 y
// ---------------------------------------------------------------------------
__global__ __launch_bounds__(256) void ln_gate_kernel(
    const float* __restrict__ y, const unsigned short* __restrict__ gate,
    const float* __restrict__ g, const float* __restrict__ bta,
    unsigned short* __restrict__ yb) {
  __shared__ float red[256];
  const int tok = blockIdx.x;
  const int tid = threadIdx.x;
  const float* row = y + (size_t)tok * Cc;
  float v0 = row[tid], v1 = row[tid + 256], v2 = row[tid + 512];

  red[tid] = v0 + v1 + v2;
  __syncthreads();
  for (int off = 128; off > 0; off >>= 1) {
    if (tid < off) red[tid] += red[tid + off];
    __syncthreads();
  }
  float mu = red[0] / (float)Cc;
  __syncthreads();

  float d0 = v0 - mu, d1 = v1 - mu, d2 = v2 - mu;
  red[tid] = d0 * d0 + d1 * d1 + d2 * d2;
  __syncthreads();
  for (int off = 128; off > 0; off >>= 1) {
    if (tid < off) red[tid] += red[tid + off];
    __syncthreads();
  }
  float rs = rsqrtf(red[0] / (float)Cc + 1e-5f);

  const unsigned short* gt = gate + (size_t)tok * Cc;
  unsigned short* ob = yb + (size_t)tok * Cc;
  float dd[3] = {d0, d1, d2};
#pragma unroll
  for (int i = 0; i < 3; ++i) {
    int c = tid + i * 256;
    float val = dd[i] * rs * g[c] + bta[c];
    ob[c] = f2bf(val * bf2f(gt[c]));
  }
}

// ---------------------------------------------------------------------------
// Workspace layout (bytes, lifetime-aliased; peak ~210 MB)
// ---------------------------------------------------------------------------
static constexpr size_t SZ_W1B  = (size_t)192 * 768 * 2;
static constexpr size_t SZ_WB   = (size_t)768 * 768 * 2;
static constexpr size_t SZ_XB   = (size_t)Mtok * Cc * 2;
static constexpr size_t SZ_XF   = (size_t)Mtok * Cc * 4;
static constexpr size_t SZ_HH   = (size_t)Mtok * 192 * 4;
static constexpr size_t SZ_WGT  = (size_t)Mtok * 9 * 4;

static constexpr size_t O_W1B  = 0;
static constexpr size_t O_WKB  = O_W1B + SZ_W1B;
static constexpr size_t O_WVB  = O_WKB + SZ_WB;
static constexpr size_t O_WRB  = O_WVB + SZ_WB;
static constexpr size_t O_WOB  = O_WRB + SZ_WB;
static constexpr size_t O_BIAS = O_WOB + SZ_WB;
static constexpr size_t O_WGT  = O_BIAS + 1024;
static constexpr size_t O_XB   = O_WGT + SZ_WGT;   // xb (phase1); SRB aliases later
static constexpr size_t O_SRB  = O_XB;
static constexpr size_t O_HH   = O_XB + SZ_XB;
static constexpr size_t O_XKB  = O_HH + SZ_HH;     // xk bf16; YB aliases later
static constexpr size_t O_YB   = O_XKB;
static constexpr size_t O_XVB  = O_XKB + SZ_XB;    // xv bf16; YF spans XVB+XRB
static constexpr size_t O_YF   = O_XVB;
static constexpr size_t O_XRB  = O_XVB + SZ_XB;
static constexpr size_t O_KB   = O_XRB + SZ_XB;
static constexpr size_t O_VB   = O_KB + SZ_XF;

extern "C" void kernel_launch(void* const* d_in, const int* in_sizes, int n_in,
                              void* d_out, int out_size, void* d_ws, size_t ws_size,
                              hipStream_t stream) {
  const float* x     = (const float*)d_in[0];
  const float* Wk    = (const float*)d_in[1];
  const float* Wv    = (const float*)d_in[2];
  const float* Wr    = (const float*)d_in[3];
  const float* Wo    = (const float*)d_in[4];
  const float* ln_g  = (const float*)d_in[5];
  const float* ln_b  = (const float*)d_in[6];
  const float* mix_k = (const float*)d_in[7];
  const float* mix_v = (const float*)d_in[8];
  const float* mix_r = (const float*)d_in[9];
  const float* decay = (const float*)d_in[10];
  const float* first = (const float*)d_in[11];
  const float* sk_w1 = (const float*)d_in[12];
  const float* sk_b1 = (const float*)d_in[13];
  const float* sk_w2 = (const float*)d_in[14];
  const float* sk_b2 = (const float*)d_in[15];
  const float* sv_w1 = (const float*)d_in[16];
  const float* sv_b1 = (const float*)d_in[17];
  const float* sv_w2 = (const float*)d_in[18];
  const float* sv_b2 = (const float*)d_in[19];
  const float* sr_w1 = (const float*)d_in[20];
  const float* sr_b1 = (const float*)d_in[21];
  const float* sr_w2 = (const float*)d_in[22];
  const float* sr_b2 = (const float*)d_in[23];

  char* ws = (char*)d_ws;
  unsigned short* W1B = (unsigned short*)(ws + O_W1B);
  unsigned short* WKB = (unsigned short*)(ws + O_WKB);
  unsigned short* WVB = (unsigned short*)(ws + O_WVB);
  unsigned short* WRB = (unsigned short*)(ws + O_WRB);
  unsigned short* WOB = (unsigned short*)(ws + O_WOB);
  float*          BS  = (float*)(ws + O_BIAS);
  float*          WGT = (float*)(ws + O_WGT);
  unsigned short* XB  = (unsigned short*)(ws + O_XB);
  unsigned short* SRB = (unsigned short*)(ws + O_SRB);
  float*          HHb = (float*)(ws + O_HH);
  unsigned short* XKB = (unsigned short*)(ws + O_XKB);
  unsigned short* XVB = (unsigned short*)(ws + O_XVB);
  unsigned short* XRB = (unsigned short*)(ws + O_XRB);
  float*          KB  = (float*)(ws + O_KB);
  float*          VB  = (float*)(ws + O_VB);
  float*          YF  = (float*)(ws + O_YF);
  unsigned short* YB  = (unsigned short*)(ws + O_YB);

  auto cvt = [&](const float* src, unsigned short* dst, int n) {
    cvt_f32_bf16_kernel<<<(n + 255) / 256, 256, 0, stream>>>(src, dst, n);
  };

  // Phase 0: bf16 conversions
  cvt(x, XB, Mtok * Cc);
  cvt(sk_w1, W1B + 0 * 64 * Cc, 64 * Cc);
  cvt(sv_w1, W1B + 1 * 64 * Cc, 64 * Cc);
  cvt(sr_w1, W1B + 2 * 64 * Cc, 64 * Cc);
  cvt(Wk, WKB, Cc * Cc);
  cvt(Wv, WVB, Cc * Cc);
  cvt(Wr, WRB, Cc * Cc);
  cvt(Wo, WOB, Cc * Cc);
  concat_bias_kernel<<<1, 64, 0, stream>>>(sk_b1, sv_b1, sr_b1, BS);

  // Phase 1: sema hidden h = relu(x @ w1^T + b1)  -> [16384, 192]
  {
    int blocks = (Mtok / 256) * (192 / 64);     // 192
    gemm_bf16_wmma_kernel<1><<<blocks, 256, 0, stream>>>(
        XB, W1B, BS, (void*)HHb, Mtok, 192, Cc);
  }

  // Phase 2: softmax sema weights -> [16384, 9]
  sema_softmax_kernel<<<(Mtok * 3 + 255) / 256, 256, 0, stream>>>(
      HHb, sk_w2, sk_b2, sv_w2, sv_b2, sr_w2, sr_b2, WGT);

  // Phase 2.5: shift + mix -> bf16 xk/xv/xr
  shiftmix_kernel<<<(int)(((size_t)Mtok * Cc + 255) / 256), 256, 0, stream>>>(
      x, WGT, mix_k, mix_v, mix_r, XKB, XVB, XRB);

  // Phase 3: projection GEMMs
  {
    int blocks = (Mtok / 256) * (Cc / 64);      // 768
    gemm_bf16_wmma_kernel<0><<<blocks, 256, 0, stream>>>(
        XKB, WKB, nullptr, (void*)KB, Mtok, Cc, Cc);
    gemm_bf16_wmma_kernel<0><<<blocks, 256, 0, stream>>>(
        XVB, WVB, nullptr, (void*)VB, Mtok, Cc, Cc);
    gemm_bf16_wmma_kernel<2><<<blocks, 256, 0, stream>>>(
        XRB, WRB, nullptr, (void*)SRB, Mtok, Cc, Cc);
  }

  // Phase 4: WKV recurrence (sequential over T)
  wkv_kernel<<<(Bb * Cc + 255) / 256, 256, 0, stream>>>(KB, VB, decay, first, YF);

  // Phase 5: LayerNorm + receptance gate -> bf16 y
  ln_gate_kernel<<<Mtok, 256, 0, stream>>>(YF, SRB, ln_g, ln_b, YB);

  // Phase 6: output projection -> d_out (fp32)
  {
    int blocks = (Mtok / 256) * (Cc / 64);      // 768
    gemm_bf16_wmma_kernel<0><<<blocks, 256, 0, stream>>>(
        YB, WOB, nullptr, d_out, Mtok, Cc, Cc);
  }
  (void)in_sizes; (void)n_in; (void)out_size; (void)ws_size;
}
